// Patch_Attention_4_41403484733635
// MI455X (gfx1250) — matrix operand order repaired and checked by reading, not compiled, not run on hardware
//
#include <hip/hip_runtime.h>

typedef __bf16 bf16_t;
typedef bf16_t v16bf __attribute__((ext_vector_type(16)));
typedef float  v8f   __attribute__((ext_vector_type(8)));

// bit-cast container: v16bf elements 0..7 live in q[0], 8..15 in q[1]
union frag_u {
    v16bf v;
    uint4 q[2];
};

// ---------------------------------------------------------------------------
// helpers
// ---------------------------------------------------------------------------
__device__ __forceinline__ int reflect64(int i) {
    return i < 0 ? -i : (i > 63 ? 126 - i : i);
}

// ---------------------------------------------------------------------------
// 0) zero fill (uint32 granularity)
// ---------------------------------------------------------------------------
__global__ void fill_zero_u32(uint32_t* __restrict__ p, long long n) {
    long long i = (long long)blockIdx.x * 256 + threadIdx.x;
    if (i < n) p[i] = 0u;
}

// ---------------------------------------------------------------------------
// 1) reflect-pad z -> bf16 padded [b*512][66 rows][72]   (row stride 72 = 144B)
// ---------------------------------------------------------------------------
__global__ void pad_reflect_kernel(const float* __restrict__ z, bf16_t* __restrict__ P) {
    long long tid = (long long)blockIdx.x * 256 + threadIdx.x;
    if (tid >= (long long)16 * 512 * 4752) return;
    int i  = (int)(tid % 4752);
    int bc = (int)(tid / 4752);
    int py = i / 72, px = i % 72;
    float v = 0.0f;
    if (px < 66) {
        int y = reflect64(py - 1), x = reflect64(px - 1);
        v = z[(size_t)bc * 4096 + y * 64 + x];
    }
    P[tid] = (bf16_t)v;
}

// ---------------------------------------------------------------------------
// 2) weight convert: OIHW fp32 -> TRANSPOSED bf16 wBt[n][k], k = tap*Cin+cin
// ---------------------------------------------------------------------------
__global__ void wcvt512_kernel(const float* __restrict__ w, bf16_t* __restrict__ wBt) {
    int tid = blockIdx.x * 256 + threadIdx.x;
    if (tid >= 512 * 4608) return;
    int k = tid % 4608;
    int n = tid / 4608;
    int cin = k & 511;
    int tap = k >> 9;             // tap = ty*3+tx
    wBt[tid] = (bf16_t)w[((size_t)n * 512 + cin) * 9 + tap];
}

__global__ void wcvtf_kernel(const float* __restrict__ wf, bf16_t* __restrict__ wBt) {
    int tid = blockIdx.x * 256 + threadIdx.x;
    if (tid >= 512 * 9504) return;
    int k = tid % 9504;
    int n = tid / 9504;
    int cin = k % 1056;
    int tap = k / 1056;
    float v = (cin < 1025) ? wf[((size_t)n * 1025 + cin) * 9 + tap] : 0.0f;
    wBt[tid] = (bf16_t)v;
}

// ---------------------------------------------------------------------------
// 3) implicit-GEMM 3x3 conv via WMMA bf16 (f32 accumulate)
//    input : padded bf16 [16*CIN][66][72]
//    weight: bf16 wBt[512][9*CIN]   (k contiguous per out-channel row)
//    out   : fp32 [16][512][4096]  (+bias)
//    block: 256 thr (8 waves). Tile M=512 spatial (8 rows x 64), N=64.
//    Per 32-ch chunk: stage 10 rows x 66 px x 32 ch to LDS [row][px][ch]
//    (reused by all 9 taps). Wave w owns output row yb*8+w: 4 m-subtiles
//    (x chunks of 16) x 4 n-subtiles = 16 WMMAs per tap.
// ---------------------------------------------------------------------------
template <int CIN>
__global__ void conv_gemm_kernel(const bf16_t* __restrict__ inP,
                                 const bf16_t* __restrict__ wBt,
                                 const float*  __restrict__ bias,
                                 float*        __restrict__ out) {
    constexpr int KTOT = 9 * CIN;
    const int PXS = 40;   // LDS px stride (halves): 80B rows -> 16B aligned, conflict-free
    __shared__ __attribute__((aligned(16))) bf16_t lsIn[10 * 66 * PXS];   // 52,800 B

    const int blk = blockIdx.x;
    const int b   = blk >> 6;          // 64 blocks per batch
    const int rem = blk & 63;
    const int yb  = rem >> 3;          // output rows yb*8 .. yb*8+7
    const int n0  = (rem & 7) << 6;    // 0..448

    const int t    = threadIdx.x;
    const int wv   = t >> 5;           // wave owns output row yb*8+wv
    const int lane = t & 31;
    const int g    = lane >> 4;
    const int mr   = lane & 15;
    const int nl   = lane & 15;

    // per-subtile weight row pointers (include lane's k-group offset g*16)
    const bf16_t* bptr[4];
#pragma unroll
    for (int j = 0; j < 4; ++j)
        bptr[j] = wBt + (size_t)(n0 + j * 16 + nl) * KTOT + (g << 4);

    v8f acc[4][4];
#pragma unroll
    for (int ms = 0; ms < 4; ++ms)
#pragma unroll
        for (int j = 0; j < 4; ++j)
#pragma unroll
            for (int v = 0; v < 8; ++v) acc[ms][j][v] = 0.0f;

    for (int c0 = 0; c0 < CIN; c0 += 32) {
        __syncthreads();
        // ---- stage A: 10 rows x 32 ch, full 72-half padded rows (9x b128),
        //      transpose-store 66 px to LDS [row][px][ch] ----
        for (int job = t; job < 320; job += 256) {
            const int rr = job >> 5;
            const int ch = job & 31;
            const uint4* src = reinterpret_cast<const uint4*>(
                inP + (size_t)(b * CIN + c0 + ch) * 4752 + (yb * 8 + rr) * 72);
            union { uint4 q[9]; bf16_t h[72]; } buf;
#pragma unroll
            for (int i = 0; i < 9; ++i) buf.q[i] = src[i];
            const int base = rr * 66 * PXS + ch;
#pragma unroll
            for (int p = 0; p < 66; ++p)
                lsIn[base + p * PXS] = buf.h[p];
        }
        __syncthreads();

        for (int tap = 0; tap < 9; ++tap) {
            const int ty = tap / 3, tx = tap % 3;
            const int kbase = tap * CIN + c0;
            // B fragments: reused across 4 m-subtiles
            frag_u fb[4];
#pragma unroll
            for (int j = 0; j < 4; ++j) {
                fb[j].q[0] = *reinterpret_cast<const uint4*>(bptr[j] + kbase);
                fb[j].q[1] = *reinterpret_cast<const uint4*>(bptr[j] + kbase + 8);
            }
            const int rowbase = (wv + ty) * 66;
#pragma unroll
            for (int ms = 0; ms < 4; ++ms) {
                const int px = ms * 16 + mr + tx;
                const bf16_t* ap = &lsIn[(rowbase + px) * PXS];
                frag_u fa;
                fa.q[0] = *reinterpret_cast<const uint4*>(ap + (g << 4));
                fa.q[1] = *reinterpret_cast<const uint4*>(ap + 8 + (g << 4));
#pragma unroll
                for (int j = 0; j < 4; ++j)
                    acc[ms][j] = __builtin_amdgcn_wmma_f32_16x16x32_bf16(
                        false, fa.v, false, fb[j].v, (short)0, acc[ms][j], false, false);
            }
        }
    }

    // ---- epilogue: bias + vectorized fp32 stores ----
    const int orow = yb * 8 + wv;
#pragma unroll
    for (int ms = 0; ms < 4; ++ms) {
        const int spx = ms * 16 + (g << 3);
#pragma unroll
        for (int j = 0; j < 4; ++j) {
            const int n  = n0 + j * 16 + nl;
            const float bv = bias[n];
            float* po = out + ((size_t)(b * 512 + n)) * 4096 + orow * 64 + spx;
            float4 o0 = make_float4(acc[ms][j][0] + bv, acc[ms][j][1] + bv,
                                    acc[ms][j][2] + bv, acc[ms][j][3] + bv);
            float4 o1 = make_float4(acc[ms][j][4] + bv, acc[ms][j][5] + bv,
                                    acc[ms][j][6] + bv, acc[ms][j][7] + bv);
            *reinterpret_cast<float4*>(po)     = o0;
            *reinterpret_cast<float4*>(po + 4) = o1;
        }
    }
}

// ---------------------------------------------------------------------------
// 4) InstanceNorm (+ReLU) then reflect-pad to bf16 (stride-72 rows)
// ---------------------------------------------------------------------------
__global__ void in_relu_pad_kernel(const float* __restrict__ conv, bf16_t* __restrict__ P) {
    const int bc = blockIdx.x;
    const float* plane = conv + (size_t)bc * 4096;
    __shared__ float red[256];
    float s = 0.f, s2 = 0.f;
    for (int i = threadIdx.x; i < 4096; i += 256) {
        float v = plane[i]; s += v; s2 += v * v;
    }
    red[threadIdx.x] = s; __syncthreads();
    for (int o = 128; o > 0; o >>= 1) { if ((int)threadIdx.x < o) red[threadIdx.x] += red[threadIdx.x + o]; __syncthreads(); }
    float mean = red[0] * (1.f / 4096.f); __syncthreads();
    red[threadIdx.x] = s2; __syncthreads();
    for (int o = 128; o > 0; o >>= 1) { if ((int)threadIdx.x < o) red[threadIdx.x] += red[threadIdx.x + o]; __syncthreads(); }
    float var  = red[0] * (1.f / 4096.f) - mean * mean;
    float rstd = rsqrtf(var + 1e-5f);
    bf16_t* op = P + (size_t)bc * 4752;
    for (int i = threadIdx.x; i < 4752; i += 256) {
        int py = i / 72, px = i % 72;
        float v = 0.f;
        if (px < 66) {
            v = (plane[reflect64(py - 1) * 64 + reflect64(px - 1)] - mean) * rstd;
            v = v > 0.f ? v : 0.f;
        }
        op[i] = (bf16_t)v;
    }
}

// ---------------------------------------------------------------------------
// 5) InstanceNorm + residual: X = z + IN(conv); also write z into concat buf
// ---------------------------------------------------------------------------
__global__ void in_residual_kernel(const float* __restrict__ conv, const float* __restrict__ z,
                                   float* __restrict__ X, bf16_t* __restrict__ catp) {
    const int bc = blockIdx.x;
    const int b = bc >> 9, c = bc & 511;
    const float* plane = conv + (size_t)bc * 4096;
    const float* zp    = z    + (size_t)bc * 4096;
    __shared__ float red[256];
    float s = 0.f, s2 = 0.f;
    for (int i = threadIdx.x; i < 4096; i += 256) {
        float v = plane[i]; s += v; s2 += v * v;
    }
    red[threadIdx.x] = s; __syncthreads();
    for (int o = 128; o > 0; o >>= 1) { if ((int)threadIdx.x < o) red[threadIdx.x] += red[threadIdx.x + o]; __syncthreads(); }
    float mean = red[0] * (1.f / 4096.f); __syncthreads();
    red[threadIdx.x] = s2; __syncthreads();
    for (int o = 128; o > 0; o >>= 1) { if ((int)threadIdx.x < o) red[threadIdx.x] += red[threadIdx.x + o]; __syncthreads(); }
    float var  = red[0] * (1.f / 4096.f) - mean * mean;
    float rstd = rsqrtf(var + 1e-5f);
    float*  xp = X + (size_t)bc * 4096;
    bf16_t* cp = catp + ((size_t)(b * 1056 + c)) * 4752;
    for (int i = threadIdx.x; i < 4096; i += 256) {
        float zv = zp[i];
        xp[i] = zv + (plane[i] - mean) * rstd;
        int y = i >> 6, x = i & 63;
        cp[(y + 1) * 72 + (x + 1)] = (bf16_t)zv;
    }
}

// ---------------------------------------------------------------------------
// 6) mask bilinear 256->64 + binarize; write channel 1024 of concat buffer
// ---------------------------------------------------------------------------
__global__ void mask_resize_kernel(const float* __restrict__ M, float* __restrict__ m64,
                                   bf16_t* __restrict__ catp) {
    int tid = blockIdx.x * 256 + threadIdx.x;   // 16*4096
    int b = tid >> 12, i = tid & 4095;
    int y = i >> 6, x = i & 63;
    const float* mb = M + (size_t)b * 65536;
    float s = mb[(4 * y + 1) * 256 + 4 * x + 1] + mb[(4 * y + 1) * 256 + 4 * x + 2] +
              mb[(4 * y + 2) * 256 + 4 * x + 1] + mb[(4 * y + 2) * 256 + 4 * x + 2];
    float m = 0.25f * s;
    if (m > 0.f) m = 1.f;
    m64[tid] = m;
    catp[((size_t)(b * 1056 + 1024)) * 4752 + (y + 1) * 72 + (x + 1)] = (bf16_t)m;
}

// 7) per-patch non-mask flag: mean of 4x4 > 0.6
__global__ void patch_mask_kernel(const float* __restrict__ m64, float* __restrict__ nonmask) {
    int b = blockIdx.x, L = threadIdx.x;
    int ph = L >> 4, pw = L & 15;
    float s = 0.f;
#pragma unroll
    for (int ki = 0; ki < 4; ++ki)
#pragma unroll
        for (int kj = 0; kj < 4; ++kj)
            s += m64[b * 4096 + (ph * 4 + ki) * 64 + pw * 4 + kj];
    nonmask[b * 256 + L] = (s > 9.6f) ? 1.f : 0.f;
}

// ---------------------------------------------------------------------------
// 8) unfold X -> xu bf16 TRANSPOSED [b][L=256][d=8192]
// ---------------------------------------------------------------------------
__global__ void unfold_kernel(const float* __restrict__ X, bf16_t* __restrict__ xut) {
    long long tid = (long long)blockIdx.x * 256 + threadIdx.x;   // 2^25 total
    int b   = (int)(tid >> 21);
    int rem = (int)(tid & 2097151);
    int L = rem >> 13, d = rem & 8191;
    int c = d >> 4, tap = d & 15;
    int y = ((L >> 4) << 2) + (tap >> 2);
    int x = ((L & 15) << 2) + (tap & 3);
    xut[tid] = (bf16_t)X[((size_t)(b * 512 + c)) * 4096 + y * 64 + x];
}

// 9) fp32 patch L2 norms
__global__ void norms_kernel(const float* __restrict__ X, float* __restrict__ norms) {
    int bl = blockIdx.x;                       // b*256 + L
    int b = bl >> 8, L = bl & 255;
    int ph = L >> 4, pw = L & 15;
    __shared__ float red[128];
    float acc = 0.f;
    for (int c = threadIdx.x; c < 512; c += 128) {
        const float* base = X + ((size_t)(b * 512 + c)) * 4096 + (ph * 4) * 64 + pw * 4;
#pragma unroll
        for (int ki = 0; ki < 4; ++ki) {
            float4 v = *(const float4*)(base + ki * 64);
            acc += v.x * v.x + v.y * v.y + v.z * v.z + v.w * v.w;
        }
    }
    red[threadIdx.x] = acc; __syncthreads();
    for (int o = 64; o > 0; o >>= 1) { if ((int)threadIdx.x < o) red[threadIdx.x] += red[threadIdx.x + o]; __syncthreads(); }
    if (threadIdx.x == 0) norms[bl] = fmaxf(sqrtf(red[0]), 1e-12f);
}

// ---------------------------------------------------------------------------
// 10) Gram matrix G = xu^T * xu via WMMA bf16 (per batch 256x256, K=8192)
//     xut transposed layout -> fragments are direct 128-bit global loads.
// ---------------------------------------------------------------------------
__global__ void gram_kernel(const bf16_t* __restrict__ xut, float* __restrict__ G) {
    const int blk = blockIdx.x;
    const int b  = blk >> 4;
    const int tt = blk & 15;
    const int q0 = (tt >> 2) << 6;
    const int n0 = (tt & 3) << 6;

    const int t    = threadIdx.x;
    const int wv   = t >> 5;
    const int lane = t & 31;
    const int g    = lane >> 4;
    const int mr   = lane & 15;
    const int nl   = lane & 15;
    const int msub = wv & 3;
    const int nh   = (wv >> 2) << 1;

    const bf16_t* arow  = xut + ((size_t)(b * 256 + q0 + msub * 16 + mr)) * 8192 + (g << 4);
    const bf16_t* brow0 = xut + ((size_t)(b * 256 + n0 + (nh + 0) * 16 + nl)) * 8192 + (g << 4);
    const bf16_t* brow1 = brow0 + (size_t)16 * 8192;

    v8f acc[2];
#pragma unroll
    for (int j = 0; j < 2; ++j)
#pragma unroll
        for (int v = 0; v < 8; ++v) acc[j][v] = 0.0f;

    for (int ks = 0; ks < 256; ++ks) {
        const int k0 = ks << 5;
        frag_u fa, fb0, fb1;
        fa.q[0]  = *reinterpret_cast<const uint4*>(arow + k0);
        fa.q[1]  = *reinterpret_cast<const uint4*>(arow + k0 + 8);
        fb0.q[0] = *reinterpret_cast<const uint4*>(brow0 + k0);
        fb0.q[1] = *reinterpret_cast<const uint4*>(brow0 + k0 + 8);
        fb1.q[0] = *reinterpret_cast<const uint4*>(brow1 + k0);
        fb1.q[1] = *reinterpret_cast<const uint4*>(brow1 + k0 + 8);
        acc[0] = __builtin_amdgcn_wmma_f32_16x16x32_bf16(
            false, fa.v, false, fb0.v, (short)0, acc[0], false, false);
        acc[1] = __builtin_amdgcn_wmma_f32_16x16x32_bf16(
            false, fa.v, false, fb1.v, (short)0, acc[1], false, false);
    }

#pragma unroll
    for (int j = 0; j < 2; ++j) {
        const int kcol = n0 + (nh + j) * 16 + nl;
#pragma unroll
        for (int v = 0; v < 8; ++v) {
            const int q = q0 + msub * 16 + (g << 3) + v;
            G[((size_t)b * 256 + q) * 256 + kcol] = acc[j][v];
        }
    }
}

// 11) masked argmax over keys (softmax is monotone -> argmax of masked corr)
__global__ void argmax_kernel(const float* __restrict__ G, const float* __restrict__ norms,
                              const float* __restrict__ nonmask, int* __restrict__ marg) {
    int b = blockIdx.x, q = threadIdx.x;
    float nq = norms[b * 256 + q];
    const float* g = G + ((size_t)b * 256 + q) * 256;
    float best = -3.4e38f;
    int bk = 0;
    for (int k = 0; k < 256; ++k) {
        float v = (nonmask[b * 256 + k] != 0.f) ? -1e9f : g[k] / (nq * norms[b * 256 + k]);
        if (v > best) { best = v; bk = k; }
    }
    marg[b * 256 + q] = bk;
}

// 12) hard-compose gather + fold, write channels 512..1023 of concat buffer
__global__ void compose_kernel(const float* __restrict__ X, const int* __restrict__ marg,
                               bf16_t* __restrict__ catp) {
    long long tid = (long long)blockIdx.x * 256 + threadIdx.x;   // 2^25 total
    int b   = (int)(tid >> 21);
    int rem = (int)(tid & 2097151);
    int c = rem >> 12, i = rem & 4095;
    int y = i >> 6, x = i & 63;
    int L = ((y >> 2) << 4) + (x >> 2);
    int a = marg[b * 256 + L];
    int sy = ((a >> 4) << 2) + (y & 3);
    int sx = ((a & 15) << 2) + (x & 3);
    float v = X[((size_t)(b * 512 + c)) * 4096 + sy * 64 + sx];
    catp[((size_t)(b * 1056 + 512 + c)) * 4752 + (y + 1) * 72 + (x + 1)] = (bf16_t)v;
}

// ---------------------------------------------------------------------------
// launcher
// ---------------------------------------------------------------------------
extern "C" void kernel_launch(void* const* d_in, const int* in_sizes, int n_in,
                              void* d_out, int out_size, void* d_ws, size_t ws_size,
                              hipStream_t stream) {
    (void)in_sizes; (void)n_in; (void)out_size; (void)ws_size;

    const float* z    = (const float*)d_in[0];
    const float* mask = (const float*)d_in[1];
    const float* w1   = (const float*)d_in[2];
    const float* b1   = (const float*)d_in[3];
    const float* w2   = (const float*)d_in[4];
    const float* b2   = (const float*)d_in[5];
    const float* wf   = (const float*)d_in[6];
    const float* bf   = (const float*)d_in[7];
    float* out        = (float*)d_out;

    char* ws = (char*)d_ws;
    // workspace layout (bytes)
    const size_t OFF_P    = 0;                               // bf16 [16*512][66][72]  : 77,856,768
    const size_t OFF_CONV = 77856768;                        // conv out fp32 / xu,G   : 134,217,728
    const size_t OFF_XU   = OFF_CONV;                        // bf16 [16][256][8192]   : 67,108,864
    const size_t OFF_G    = OFF_CONV + 67108864;             // fp32 [16][256][256]    : 4,194,304
    const size_t OFF_X    = OFF_CONV + 134217728;            // fp32 [16][512][4096]   : 134,217,728
    const size_t OFF_CAT  = OFF_X + 134217728;               // bf16 [16*1056][66][72] : 160,579,584
    const size_t OFF_W1   = OFF_CAT + 160579584;             // bf16 [512][4608]
    const size_t OFF_W2   = OFF_W1 + 4718592;
    const size_t OFF_WF   = OFF_W2 + 4718592;                // bf16 [512][9504]
    const size_t OFF_M64  = OFF_WF + 9732096;                // fp32 [16][4096]
    const size_t OFF_NRM  = OFF_M64 + 262144;                // fp32 [16][256]
    const size_t OFF_NMK  = OFF_NRM + 16384;                 // fp32 [16][256]
    const size_t OFF_ARG  = OFF_NMK + 16384;                 // int  [16][256]

    bf16_t* P    = (bf16_t*)(ws + OFF_P);
    float*  CONV = (float*)(ws + OFF_CONV);
    bf16_t* XU   = (bf16_t*)(ws + OFF_XU);
    float*  G    = (float*)(ws + OFF_G);
    float*  X    = (float*)(ws + OFF_X);
    bf16_t* CAT  = (bf16_t*)(ws + OFF_CAT);
    bf16_t* W1B  = (bf16_t*)(ws + OFF_W1);
    bf16_t* W2B  = (bf16_t*)(ws + OFF_W2);
    bf16_t* WFB  = (bf16_t*)(ws + OFF_WF);
    float*  M64  = (float*)(ws + OFF_M64);
    float*  NRM  = (float*)(ws + OFF_NRM);
    float*  NMK  = (float*)(ws + OFF_NMK);
    int*    ARG  = (int*)(ws + OFF_ARG);

    // 0) zero concat buffer (borders + pad channels + tail lanes must be zero)
    {
        long long n32 = (long long)16 * 1056 * 4752 / 2;     // 40,144,896
        fill_zero_u32<<<(int)((n32 + 255) / 256), 256, 0, stream>>>((uint32_t*)CAT, n32);
    }
    // 1) pad z
    pad_reflect_kernel<<<152064, 256, 0, stream>>>(z, P);
    // 2) weights (transposed [n][k])
    wcvt512_kernel<<<9216, 256, 0, stream>>>(w1, W1B);
    wcvt512_kernel<<<9216, 256, 0, stream>>>(w2, W2B);
    wcvtf_kernel<<<19008, 256, 0, stream>>>(wf, WFB);
    // 3) conv1
    conv_gemm_kernel<512><<<1024, 256, 0, stream>>>(P, W1B, b1, CONV);
    // 4) IN + ReLU + re-pad (overwrites P)
    in_relu_pad_kernel<<<8192, 256, 0, stream>>>(CONV, P);
    // 5) conv2
    conv_gemm_kernel<512><<<1024, 256, 0, stream>>>(P, W2B, b2, CONV);
    // 6) IN + residual -> X ; z into concat channels [0,512)
    in_residual_kernel<<<8192, 256, 0, stream>>>(CONV, z, X, CAT);
    // 7) mask resize + binarize ; channel 1024
    mask_resize_kernel<<<256, 256, 0, stream>>>(mask, M64, CAT);
    patch_mask_kernel<<<16, 256, 0, stream>>>(M64, NMK);
    // 8) unfold (transposed) + norms (CONV region is dead; XU/G alias it)
    unfold_kernel<<<131072, 256, 0, stream>>>(X, XU);
    norms_kernel<<<4096, 128, 0, stream>>>(X, NRM);
    // 9) Gram + argmax
    gram_kernel<<<256, 256, 0, stream>>>(XU, G);
    argmax_kernel<<<16, 256, 0, stream>>>(G, NRM, NMK, ARG);
    // 10) compose into concat channels [512,1024)
    compose_kernel<<<131072, 256, 0, stream>>>(X, ARG, CAT);
    // 11) final conv (zero padding already in CAT borders) -> d_out
    conv_gemm_kernel<1056><<<1024, 256, 0, stream>>>(CAT, WFB, bf, out);
}
